// EdgeNet_40656160424625
// MI455X (gfx1250) — matrix operand order repaired and checked by reading, not compiled, not run on hardware
//
#include <hip/hip_runtime.h>
#include <hip/hip_bf16.h>

// ---------------------------------------------------------------------------
// DGCNN (EdgeConv x4 + mlp5 + global max pool + 2 FC + head) for gfx1250.
// GEMMs: v_wmma_f32_16x16x32_f16 (wave32 WMMA, f32 accumulate).
// Data movement: global_load_async_to_lds_b128 (ASYNCcnt DMA) double-buffered
// against WMMA consumption. Each wave owns TPW consecutive N-tiles of a single
// M-subtile, so one A fragment per K-chunk is register-resident and reused.
// ---------------------------------------------------------------------------

#define BATCH 8
#define NPTS  2048
#define BNW   (BATCH * NPTS)   // 16384 nodes total
#define KNN_K 16
#define EPSBN 1e-5f

typedef __attribute__((ext_vector_type(16))) _Float16 v16h;
typedef __attribute__((ext_vector_type(8)))  _Float16 v8h;
typedef __attribute__((ext_vector_type(8)))  float    v8f;

__device__ __forceinline__ v8f zero8() {
  v8f r;
#pragma unroll
  for (int i = 0; i < 8; ++i) r[i] = 0.f;
  return r;
}

__device__ __forceinline__ v16h mk16(v8h lo, v8h hi) {
  v16h r;
#pragma unroll
  for (int i = 0; i < 8; ++i) { r[i] = lo[i]; r[i + 8] = hi[i]; }
  return r;
}

// A fragment (16x32 f16). Lane L holds row M = L%16.
// ISA layout: lanes 0-15 -> K = {k0+0..7, k0+16..23}; lanes 16-31 -> +8.
__device__ __forceinline__ v16h load_a_frag(const _Float16* row, int k0, int lane) {
  int off = (lane & 16) ? 8 : 0;
  v8h lo = *(const v8h*)(row + k0 + off);
  v8h hi = *(const v8h*)(row + k0 + off + 16);
  return mk16(lo, hi);
}

// B fragment (32x16 f16). Lane L holds column N = L%16 (column-major staged
// rows): lanes 0-15 -> K = {k0+0..7, k0+16..23}; lanes 16-31 -> +8.
__device__ __forceinline__ v16h load_b_frag(const _Float16* row, int k0, int lane) {
  int off = (lane & 16) ? 8 : 0;
  v8h lo = *(const v8h*)(row + k0 + off);
  v8h hi = *(const v8h*)(row + k0 + off + 16);
  return mk16(lo, hi);
}

// ---------------------------------------------------------------------------
// Async global->LDS DMA (ASYNCcnt). Per-lane 16B copy; inline asm because the
// clang builtins don't cover GLOBAL_LOAD_ASYNC_TO_LDS on this toolchain.
// Generic pointers into LDS carry the LDS byte offset in the low 32 bits
// (ISA: LDS aperture maps addr[31:0]).
// ---------------------------------------------------------------------------
__device__ __forceinline__ uint32_t lds_addr32(const void* p) {
  return (uint32_t)(uintptr_t)p;
}
__device__ __forceinline__ void async_copy_b128(uint32_t lds_byte_addr,
                                                const void* gptr) {
  asm volatile("global_load_async_to_lds_b128 %0, %1, off"
               :: "v"(lds_byte_addr), "v"((unsigned long long)(uintptr_t)gptr)
               : "memory");
}
__device__ __forceinline__ void wait_async0() {
  asm volatile("s_wait_asynccnt 0x0" ::: "memory");
}

// ---------------------------------------------------------------------------
// Workspace layout (bytes)
// ---------------------------------------------------------------------------
static constexpr size_t OFF_WT1  = 0;                                   // [64][32]   f16
static constexpr size_t OFF_WT2  = OFF_WT1  + (size_t)64 * 32 * 2;      // [64][128]  f16
static constexpr size_t OFF_WT3  = OFF_WT2  + (size_t)64 * 128 * 2;     // [128][128] f16
static constexpr size_t OFF_WT4  = OFF_WT3  + (size_t)128 * 128 * 2;    // [256][256] f16
static constexpr size_t OFF_WT5  = OFF_WT4  + (size_t)256 * 256 * 2;    // [1024][512] f16
static constexpr size_t OFF_X0   = OFF_WT5  + (size_t)1024 * 512 * 2;   // [BN][6]   f32
static constexpr size_t OFF_X1   = OFF_X0   + (size_t)BNW * 6 * 4;      // [BN][64]  f32
static constexpr size_t OFF_X2   = OFF_X1   + (size_t)BNW * 64 * 4;     // [BN][64]  f32
static constexpr size_t OFF_X3   = OFF_X2   + (size_t)BNW * 64 * 4;     // [BN][128] f32
static constexpr size_t OFF_X4   = OFF_X3   + (size_t)BNW * 128 * 4;    // [BN][256] f32
static constexpr size_t OFF_XH   = OFF_X4   + (size_t)BNW * 256 * 4;    // [BN][<=128] f16 (reused)
static constexpr size_t OFF_SQ   = OFF_XH   + (size_t)BNW * 128 * 2;    // [BN] f32
static constexpr size_t OFF_KNN  = OFF_SQ   + (size_t)BNW * 4;          // [BN][16] int
static constexpr size_t OFF_ZMAX = OFF_KNN  + (size_t)BNW * KNN_K * 4;  // [BN][<=256] f32
static constexpr size_t OFF_ZMIN = OFF_ZMAX + (size_t)BNW * 256 * 4;    // [BN][<=256] f32
static constexpr size_t OFF_SUM  = OFF_ZMIN + (size_t)BNW * 256 * 4;    // [5][1024] f32
static constexpr size_t OFF_SQS  = OFF_SUM  + (size_t)5 * 1024 * 4;     // [5][1024] f32
static constexpr size_t OFF_GMAX = OFF_SQS  + (size_t)5 * 1024 * 4;     // [8][1024] f32
static constexpr size_t OFF_GMIN = OFF_GMAX + (size_t)8 * 1024 * 4;     // [8][1024] f32

// ---------------------------------------------------------------------------
// Small utility kernels
// ---------------------------------------------------------------------------
__global__ void k_init(float* sums, float* sqs, float* gmax, int* gminBits) {
  int t = blockIdx.x * blockDim.x + threadIdx.x;
  if (t < 5 * 1024) { sums[t] = 0.f; sqs[t] = 0.f; }
  if (t < 8 * 1024) { gmax[t] = 0.f; gminBits[t] = 0x7F800000; }  // +inf
}

// W f32 [kreal][cout] -> Wt f16 [cout][kpad] (transposed + zero padded K)
__global__ void k_convert_wt(const float* __restrict__ W, _Float16* __restrict__ Wt,
                             int kreal, int kpad, int cout) {
  int stride = gridDim.x * blockDim.x;
  int tot = kpad * cout;
  for (int t = blockIdx.x * blockDim.x + threadIdx.x; t < tot; t += stride) {
    int n = t / kpad, k = t % kpad;
    Wt[t] = (k < kreal) ? (_Float16)W[(size_t)k * cout + n] : (_Float16)0.f;
  }
}

__global__ void k_concat(const float* __restrict__ x, const float* __restrict__ pos,
                         float* __restrict__ X0) {
  int stride = gridDim.x * blockDim.x;
  for (int n = blockIdx.x * blockDim.x + threadIdx.x; n < BNW; n += stride) {
#pragma unroll
    for (int c = 0; c < 3; ++c) {
      X0[n * 6 + c]     = x[n * 3 + c];
      X0[n * 6 + 3 + c] = pos[n * 3 + c];
    }
  }
}

// f32 features -> f16 (zero padded to Cpad) + squared norms
__global__ void k_prep(const float* __restrict__ Xf, int C, _Float16* __restrict__ Xh,
                       int Cpad, float* __restrict__ sq) {
  int stride = gridDim.x * blockDim.x;
  for (int n = blockIdx.x * blockDim.x + threadIdx.x; n < BNW; n += stride) {
    float s = 0.f;
    for (int c = 0; c < Cpad; ++c) {
      float v = (c < C) ? Xf[(size_t)n * C + c] : 0.f;
      Xh[(size_t)n * Cpad + c] = (_Float16)v;
      if (c < C) s += v * v;
    }
    sq[n] = s;
  }
}

// ---------------------------------------------------------------------------
// kNN: per workgroup 16 rows; WMMA Gram tiles 16x16, 128-col chunks in LDS,
// fused register-resident top-16 (includes self: d_ii = 0). The 16 A-rows
// are DMA'd into LDS once (async); A fragments are hoisted into registers
// and reused across all 16 column chunks.
// ---------------------------------------------------------------------------
template <int CPAD>
__global__ __launch_bounds__(256) void k_knn(const _Float16* __restrict__ Xh,
                                             const float* __restrict__ sq,
                                             int* __restrict__ knn) {
  constexpr int NKC = CPAD / 32;
  __shared__ alignas(16) _Float16 Arows[16][CPAD];
  __shared__ float dtile[16][132];
  __shared__ float sqR[16];
  int tid = threadIdx.x, lane = tid & 31, wave = tid >> 5;
  int row0 = blockIdx.x * 16;
  int b = row0 / NPTS;
  int gbase = b * NPTS;
  if (tid < 16) sqR[tid] = sq[row0 + tid];

  // async-stage the 16 rows (CPAD halves each) -> 16B segments
  {
    constexpr int SEGS = CPAD / 8;          // 16B segments per row
    constexpr int TOT  = 16 * SEGS;
    for (int l = tid; l < TOT; l += 256) {
      int r = l / SEGS, seg = l % SEGS;
      async_copy_b128(lds_addr32(&Arows[r][0]) + seg * 16,
                      Xh + (size_t)(row0 + r) * CPAD + seg * 8);
    }
  }

  float bd[16]; int bi[16];
#pragma unroll
  for (int q = 0; q < 16; ++q) { bd[q] = 3.4e38f; bi[q] = gbase; }
  wait_async0();
  __syncthreads();

  // hoist A fragments out of the column sweep (loop-invariant)
  v16h afr[NKC];
  {
    const _Float16* arow = &Arows[lane & 15][0];
#pragma unroll
    for (int kc = 0; kc < NKC; ++kc) afr[kc] = load_a_frag(arow, kc * 32, lane);
  }

  for (int cc = 0; cc < NPTS / 128; ++cc) {
    int c0 = gbase + cc * 128 + wave * 16;
    const _Float16* brow = Xh + (size_t)(c0 + (lane & 15)) * CPAD;
    v8f acc = zero8();
#pragma unroll
    for (int kc = 0; kc < NKC; ++kc) {
      v16h bb = load_b_frag(brow, kc * 32, lane);
      acc = __builtin_amdgcn_wmma_f32_16x16x32_f16(false, afr[kc], false, bb,
                                                   (short)0, acc, false, false);
    }
    float sqB = sq[c0 + (lane & 15)];
    int moff = (lane & 16) ? 8 : 0;
#pragma unroll
    for (int e = 0; e < 8; ++e) {
      int m = e + moff;
      dtile[m][wave * 16 + (lane & 15)] = sqR[m] + sqB - 2.0f * acc[e];
    }
    __syncthreads();
    if (tid < 16) {
      for (int c = 0; c < 128; ++c) {
        float v = dtile[tid][c];
        if (v < bd[15]) {
          bd[15] = v; bi[15] = gbase + cc * 128 + c;
#pragma unroll
          for (int q = 15; q > 0; --q) {
            if (bd[q] < bd[q - 1]) {
              float tf = bd[q]; bd[q] = bd[q - 1]; bd[q - 1] = tf;
              int   ti = bi[q]; bi[q] = bi[q - 1]; bi[q - 1] = ti;
            }
          }
        }
      }
    }
    __syncthreads();
  }
  if (tid < 16) {
#pragma unroll
    for (int q = 0; q < 16; ++q) knn[(size_t)(row0 + tid) * KNN_K + q] = bi[q];
  }
}

// ---------------------------------------------------------------------------
// EdgeConv MLP: per workgroup 4 nodes = 64 edge rows (one node = one M-tile).
// Double-buffered pipeline: async W-chunk DMA + VALU A-gather for chunk kc+1
// overlap WMMA on chunk kc. Wave w owns tiles [w*TPW, (w+1)*TPW) -> a single
// M-subtile per wave (TPW divides NT), so one A frag per chunk, reused TPW x.
// Epilogue: bias+relu, channel sum/sumsq atomics, per-node max & min over k.
// ---------------------------------------------------------------------------
template <int CREAL, int KPAD, int COUT>
__global__ __launch_bounds__(256) void k_edge(const float* __restrict__ Xf,
                                              const int* __restrict__ knn,
                                              const _Float16* __restrict__ Wt,
                                              const float* __restrict__ bias,
                                              float* __restrict__ zmax,
                                              float* __restrict__ zmin,
                                              float* __restrict__ psum,
                                              float* __restrict__ psq) {
  constexpr int NT  = COUT / 16;
  constexpr int TPW = (4 * NT) / 8;
  constexpr int NK  = KPAD / 32;
  static_assert(NT % TPW == 0, "wave tiles must share one M-subtile");
  __shared__ alignas(16) _Float16 Ach[2][64][32];
  __shared__ alignas(16) _Float16 Wch[2][COUT][32];
  __shared__ int jidx[64];
  int tid = threadIdx.x, lane = tid & 31, wave = tid >> 5;
  int node0 = blockIdx.x * 4;
  int mtw = (wave * TPW) / NT;         // this wave's M-subtile
  int ntb = (wave * TPW) % NT;         // first N-subtile
  if (tid < 64) jidx[tid] = knn[(size_t)(node0 + (tid >> 4)) * KNN_K + (tid & 15)];
  __syncthreads();

  auto stageW = [&](int kc, int buf) {
    constexpr int TOT = COUT * 4;           // 16B segments
    for (int l = tid; l < TOT; l += 256) {
      int n = l >> 2, seg = l & 3;
      async_copy_b128(lds_addr32(&Wch[buf][n][0]) + seg * 16,
                      (const char*)Wt + ((size_t)n * KPAD + kc * 32) * 2 + seg * 16);
    }
  };
  auto stageA = [&](int kc, int buf) {
    for (int t = tid; t < 64 * 32; t += 256) {
      int r = t >> 5, k = t & 31, gk = kc * 32 + k;
      int i = node0 + (r >> 4);
      float v = 0.f;
      if (gk < CREAL) {
        v = Xf[(size_t)i * CREAL + gk];
      } else if (gk < 2 * CREAL) {
        int c = gk - CREAL; int j = jidx[r];
        v = Xf[(size_t)j * CREAL + c] - Xf[(size_t)i * CREAL + c];
      }
      Ach[buf][r][k] = (_Float16)v;
    }
  };

  v8f acc[TPW];
#pragma unroll
  for (int t = 0; t < TPW; ++t) acc[t] = zero8();

  stageW(0, 0);
  stageA(0, 0);
  wait_async0();
  __syncthreads();

  for (int kc = 0; kc < NK; ++kc) {
    int cur = kc & 1, nxt = cur ^ 1;
    if (kc + 1 < NK) { stageW(kc + 1, nxt); stageA(kc + 1, nxt); }
    v16h a = load_a_frag(&Ach[cur][mtw * 16 + (lane & 15)][0], 0, lane);
#pragma unroll
    for (int ti = 0; ti < TPW; ++ti) {
      v16h bb = load_b_frag(&Wch[cur][(ntb + ti) * 16 + (lane & 15)][0], 0, lane);
      acc[ti] = __builtin_amdgcn_wmma_f32_16x16x32_f16(false, a, false, bb,
                                                       (short)0, acc[ti], false, false);
    }
    wait_async0();
    __syncthreads();
  }

#pragma unroll
  for (int ti = 0; ti < TPW; ++ti) {
    int col = (ntb + ti) * 16 + (lane & 15);
    float bc = bias[col];
    float s = 0.f, q = 0.f, mx = 0.f, mn = 3.4e38f;
#pragma unroll
    for (int e = 0; e < 8; ++e) {
      float z = fmaxf(acc[ti][e] + bc, 0.f);
      s += z; q += z * z; mx = fmaxf(mx, z); mn = fminf(mn, z);
    }
    s += __shfl_xor(s, 16, 32);
    q += __shfl_xor(q, 16, 32);
    mx = fmaxf(mx, __shfl_xor(mx, 16, 32));
    mn = fminf(mn, __shfl_xor(mn, 16, 32));
    if (lane < 16) {
      atomicAdd(&psum[col], s);
      atomicAdd(&psq[col], q);
      int node = node0 + mtw;
      zmax[(size_t)node * COUT + col] = mx;
      zmin[(size_t)node * COUT + col] = mn;
    }
  }
}

// BN affine is monotone per channel: max over edges of affine(z) is
// affine(zmax) when scale>=0 else affine(zmin).
template <int COUT>
__global__ void k_finalize(const float* __restrict__ zmax, const float* __restrict__ zmin,
                           const float* __restrict__ psum, const float* __restrict__ psq,
                           const float* __restrict__ g, const float* __restrict__ be,
                           float* __restrict__ Xout) {
  const float R = (float)((size_t)BNW * KNN_K);
  int stride = gridDim.x * blockDim.x;
  size_t tot = (size_t)BNW * COUT;
  for (size_t t = blockIdx.x * blockDim.x + threadIdx.x; t < tot; t += stride) {
    int c = (int)(t % COUT);
    float m  = psum[c] / R;
    float vr = psq[c] / R - m * m;
    float sc = g[c] * rsqrtf(vr + EPSBN);
    float sh = be[c] - sc * m;
    float v  = (sc >= 0.f) ? zmax[t] : zmin[t];
    Xout[t] = sc * v + sh;
  }
}

// ---------------------------------------------------------------------------
// mlp5 (512 -> 1024) fused with global max pool: per-graph max/min via
// int-bit atomics (z >= 0 after relu so float bit order == int order).
// Grid: (BN/64 node tiles, 4 column chunks of 256). Same double-buffered
// async-W pipeline as k_edge; one A frag per wave per chunk (TPW=8, NT=16).
// ---------------------------------------------------------------------------
__global__ __launch_bounds__(256) void k_mlp5(const float* __restrict__ X1,
                                              const float* __restrict__ X2,
                                              const float* __restrict__ X3,
                                              const float* __restrict__ X4,
                                              const _Float16* __restrict__ Wt,
                                              const float* __restrict__ bias,
                                              float* __restrict__ psum,
                                              float* __restrict__ psq,
                                              int* __restrict__ gmaxBits,
                                              int* __restrict__ gminBits) {
  __shared__ alignas(16) _Float16 Ach[2][64][32];
  __shared__ alignas(16) _Float16 Wch[2][256][32];
  int tid = threadIdx.x, lane = tid & 31, wave = tid >> 5;
  int node0 = blockIdx.x * 64;
  int ncb = blockIdx.y * 256;
  int b = node0 / NPTS;
  int mtw = wave >> 1;                 // (wave*8)/16
  int ntb = (wave & 1) * 8;            // (wave*8)%16

  auto stageW = [&](int kc, int buf) {
    for (int l = tid; l < 1024; l += 256) {      // 256 rows * 4 segs
      int n = l >> 2, seg = l & 3;
      async_copy_b128(lds_addr32(&Wch[buf][n][0]) + seg * 16,
                      (const char*)Wt + ((size_t)(ncb + n) * 512 + kc * 32) * 2 + seg * 16);
    }
  };
  auto stageA = [&](int kc, int buf) {
    for (int t = tid; t < 64 * 32; t += 256) {
      int r = t >> 5, k = t & 31, gk = kc * 32 + k;
      int i = node0 + r;
      float v;
      if (gk < 64)        v = X1[(size_t)i * 64  + gk];
      else if (gk < 128)  v = X2[(size_t)i * 64  + gk - 64];
      else if (gk < 256)  v = X3[(size_t)i * 128 + gk - 128];
      else                v = X4[(size_t)i * 256 + gk - 256];
      Ach[buf][r][k] = (_Float16)v;
    }
  };

  v8f acc[8];
#pragma unroll
  for (int t = 0; t < 8; ++t) acc[t] = zero8();

  stageW(0, 0);
  stageA(0, 0);
  wait_async0();
  __syncthreads();

  for (int kc = 0; kc < 16; ++kc) {
    int cur = kc & 1, nxt = cur ^ 1;
    if (kc + 1 < 16) { stageW(kc + 1, nxt); stageA(kc + 1, nxt); }
    v16h a = load_a_frag(&Ach[cur][mtw * 16 + (lane & 15)][0], 0, lane);
#pragma unroll
    for (int ti = 0; ti < 8; ++ti) {
      v16h bb = load_b_frag(&Wch[cur][(ntb + ti) * 16 + (lane & 15)][0], 0, lane);
      acc[ti] = __builtin_amdgcn_wmma_f32_16x16x32_f16(false, a, false, bb,
                                                       (short)0, acc[ti], false, false);
    }
    wait_async0();
    __syncthreads();
  }

#pragma unroll
  for (int ti = 0; ti < 8; ++ti) {
    int col = ncb + (ntb + ti) * 16 + (lane & 15);
    float bc = bias[col];
    float s = 0.f, q = 0.f, mx = 0.f, mn = 3.4e38f;
#pragma unroll
    for (int e = 0; e < 8; ++e) {
      float z = fmaxf(acc[ti][e] + bc, 0.f);
      s += z; q += z * z; mx = fmaxf(mx, z); mn = fminf(mn, z);
    }
    s += __shfl_xor(s, 16, 32);
    q += __shfl_xor(q, 16, 32);
    mx = fmaxf(mx, __shfl_xor(mx, 16, 32));
    mn = fminf(mn, __shfl_xor(mn, 16, 32));
    if (lane < 16) {
      atomicAdd(&psum[col], s);
      atomicAdd(&psq[col], q);
      atomicMax(&gmaxBits[b * 1024 + col], __float_as_int(mx));
      atomicMin(&gminBits[b * 1024 + col], __float_as_int(mn));
    }
  }
}

// ---------------------------------------------------------------------------
// Head: mlp5 BN+pool finalize, fin1 (1024->512 + BN), fin2 (512->256 + BN),
// out (256->10). Tiny (8 rows) -> one workgroup, VALU.
// ---------------------------------------------------------------------------
__global__ __launch_bounds__(256) void k_head(const float* __restrict__ psum5,
                                              const float* __restrict__ psq5,
                                              const float* __restrict__ g5,
                                              const float* __restrict__ be5,
                                              const float* __restrict__ gmax,
                                              const float* __restrict__ gmin,
                                              const float* __restrict__ F1W,
                                              const float* __restrict__ F1b,
                                              const float* __restrict__ F1g,
                                              const float* __restrict__ F1be,
                                              const float* __restrict__ F2W,
                                              const float* __restrict__ F2b,
                                              const float* __restrict__ F2g,
                                              const float* __restrict__ F2be,
                                              const float* __restrict__ OW,
                                              const float* __restrict__ Ob,
                                              float* __restrict__ out) {
  __shared__ float gL[8][1024];
  __shared__ float t1[8][512];
  __shared__ float t2[8][256];
  int tid = threadIdx.x;
  const float Rn = (float)BNW;

  for (int t = tid; t < 8 * 1024; t += 256) {
    int c = t & 1023;
    float m  = psum5[c] / Rn;
    float vr = psq5[c] / Rn - m * m;
    float sc = g5[c] * rsqrtf(vr + EPSBN);
    float sh = be5[c] - sc * m;
    float v  = (sc >= 0.f) ? gmax[t] : gmin[t];
    gL[t >> 10][c] = sc * v + sh;
  }
  __syncthreads();

  for (int t = tid; t < 8 * 512; t += 256) {
    int bi = t >> 9, n = t & 511;
    float s = F1b[n];
    for (int k = 0; k < 1024; ++k) s += gL[bi][k] * F1W[(size_t)k * 512 + n];
    t1[bi][n] = fmaxf(s, 0.f);
  }
  __syncthreads();
  for (int n = tid; n < 512; n += 256) {
    float m = 0.f, q = 0.f;
    for (int bi = 0; bi < 8; ++bi) { float v = t1[bi][n]; m += v; q += v * v; }
    m *= 0.125f;
    float vr = q * 0.125f - m * m;
    float sc = F1g[n] * rsqrtf(vr + EPSBN), sh = F1be[n] - sc * m;
    for (int bi = 0; bi < 8; ++bi) t1[bi][n] = sc * t1[bi][n] + sh;
  }
  __syncthreads();

  for (int t = tid; t < 8 * 256; t += 256) {
    int bi = t >> 8, n = t & 255;
    float s = F2b[n];
    for (int k = 0; k < 512; ++k) s += t1[bi][k] * F2W[(size_t)k * 256 + n];
    t2[bi][n] = fmaxf(s, 0.f);
  }
  __syncthreads();
  for (int n = tid; n < 256; n += 256) {
    float m = 0.f, q = 0.f;
    for (int bi = 0; bi < 8; ++bi) { float v = t2[bi][n]; m += v; q += v * v; }
    m *= 0.125f;
    float vr = q * 0.125f - m * m;
    float sc = F2g[n] * rsqrtf(vr + EPSBN), sh = F2be[n] - sc * m;
    for (int bi = 0; bi < 8; ++bi) t2[bi][n] = sc * t2[bi][n] + sh;
  }
  __syncthreads();

  for (int t = tid; t < 80; t += 256) {
    int bi = t / 10, j = t % 10;
    float s = Ob[j];
    for (int k = 0; k < 256; ++k) s += t2[bi][k] * OW[k * 10 + j];
    out[t] = s;
  }
}

// ---------------------------------------------------------------------------
// Launch
// ---------------------------------------------------------------------------
extern "C" void kernel_launch(void* const* d_in, const int* in_sizes, int n_in,
                              void* d_out, int out_size, void* d_ws, size_t ws_size,
                              hipStream_t stream) {
  (void)in_sizes; (void)n_in; (void)out_size; (void)ws_size;
  const float* x   = (const float*)d_in[0];
  const float* pos = (const float*)d_in[1];
  const float* W1  = (const float*)d_in[2];
  const float* b1  = (const float*)d_in[3];
  const float* g1  = (const float*)d_in[4];
  const float* be1 = (const float*)d_in[5];
  const float* W2  = (const float*)d_in[6];
  const float* b2  = (const float*)d_in[7];
  const float* g2  = (const float*)d_in[8];
  const float* be2 = (const float*)d_in[9];
  const float* W3  = (const float*)d_in[10];
  const float* b3  = (const float*)d_in[11];
  const float* g3  = (const float*)d_in[12];
  const float* be3 = (const float*)d_in[13];
  const float* W4  = (const float*)d_in[14];
  const float* b4  = (const float*)d_in[15];
  const float* g4  = (const float*)d_in[16];
  const float* be4 = (const float*)d_in[17];
  const float* W5  = (const float*)d_in[18];
  const float* b5  = (const float*)d_in[19];
  const float* g5  = (const float*)d_in[20];
  const float* be5 = (const float*)d_in[21];
  const float* F1W = (const float*)d_in[22];
  const float* F1b = (const float*)d_in[23];
  const float* F1g = (const float*)d_in[24];
  const float* F1be= (const float*)d_in[25];
  const float* F2W = (const float*)d_in[26];
  const float* F2b = (const float*)d_in[27];
  const float* F2g = (const float*)d_in[28];
  const float* F2be= (const float*)d_in[29];
  const float* OW  = (const float*)d_in[30];
  const float* Ob  = (const float*)d_in[31];

  char* ws = (char*)d_ws;
  _Float16* Wt1 = (_Float16*)(ws + OFF_WT1);
  _Float16* Wt2 = (_Float16*)(ws + OFF_WT2);
  _Float16* Wt3 = (_Float16*)(ws + OFF_WT3);
  _Float16* Wt4 = (_Float16*)(ws + OFF_WT4);
  _Float16* Wt5 = (_Float16*)(ws + OFF_WT5);
  float* X0   = (float*)(ws + OFF_X0);
  float* X1   = (float*)(ws + OFF_X1);
  float* X2   = (float*)(ws + OFF_X2);
  float* X3   = (float*)(ws + OFF_X3);
  float* X4   = (float*)(ws + OFF_X4);
  _Float16* Xh = (_Float16*)(ws + OFF_XH);
  float* sq   = (float*)(ws + OFF_SQ);
  int*   knn  = (int*)(ws + OFF_KNN);
  float* zmax = (float*)(ws + OFF_ZMAX);
  float* zmin = (float*)(ws + OFF_ZMIN);
  float* sums = (float*)(ws + OFF_SUM);
  float* sqs  = (float*)(ws + OFF_SQS);
  float* gmax = (float*)(ws + OFF_GMAX);
  float* gmin = (float*)(ws + OFF_GMIN);
  float* out  = (float*)d_out;

  k_init<<<32, 256, 0, stream>>>(sums, sqs, gmax, (int*)gmin);

  k_convert_wt<<<64, 256, 0, stream>>>(W1, Wt1, 12, 32, 64);
  k_convert_wt<<<64, 256, 0, stream>>>(W2, Wt2, 128, 128, 64);
  k_convert_wt<<<64, 256, 0, stream>>>(W3, Wt3, 128, 128, 128);
  k_convert_wt<<<128, 256, 0, stream>>>(W4, Wt4, 256, 256, 256);
  k_convert_wt<<<512, 256, 0, stream>>>(W5, Wt5, 512, 512, 1024);

  k_concat<<<64, 256, 0, stream>>>(x, pos, X0);

  // ---- Layer 1: knn on [x,pos] (C=6), EdgeConv 12->64
  k_prep<<<64, 256, 0, stream>>>(X0, 6, Xh, 32, sq);
  k_knn<32><<<BNW / 16, 256, 0, stream>>>(Xh, sq, knn);
  k_edge<6, 32, 64><<<BNW / 4, 256, 0, stream>>>(X0, knn, Wt1, b1, zmax, zmin,
                                                 sums + 0 * 1024, sqs + 0 * 1024);
  k_finalize<64><<<256, 256, 0, stream>>>(zmax, zmin, sums + 0 * 1024, sqs + 0 * 1024,
                                          g1, be1, X1);

  // ---- Layer 2: knn on x1 (C=64), EdgeConv 128->64
  k_prep<<<64, 256, 0, stream>>>(X1, 64, Xh, 64, sq);
  k_knn<64><<<BNW / 16, 256, 0, stream>>>(Xh, sq, knn);
  k_edge<64, 128, 64><<<BNW / 4, 256, 0, stream>>>(X1, knn, Wt2, b2, zmax, zmin,
                                                   sums + 1 * 1024, sqs + 1 * 1024);
  k_finalize<64><<<256, 256, 0, stream>>>(zmax, zmin, sums + 1 * 1024, sqs + 1 * 1024,
                                          g2, be2, X2);

  // ---- Layer 3: knn on x2 (C=64), EdgeConv 128->128
  k_prep<<<64, 256, 0, stream>>>(X2, 64, Xh, 64, sq);
  k_knn<64><<<BNW / 16, 256, 0, stream>>>(Xh, sq, knn);
  k_edge<64, 128, 128><<<BNW / 4, 256, 0, stream>>>(X2, knn, Wt3, b3, zmax, zmin,
                                                    sums + 2 * 1024, sqs + 2 * 1024);
  k_finalize<128><<<256, 256, 0, stream>>>(zmax, zmin, sums + 2 * 1024, sqs + 2 * 1024,
                                           g3, be3, X3);

  // ---- Layer 4: knn on x3 (C=128), EdgeConv 256->256
  k_prep<<<64, 256, 0, stream>>>(X3, 128, Xh, 128, sq);
  k_knn<128><<<BNW / 16, 256, 0, stream>>>(Xh, sq, knn);
  k_edge<128, 256, 256><<<BNW / 4, 256, 0, stream>>>(X3, knn, Wt4, b4, zmax, zmin,
                                                     sums + 3 * 1024, sqs + 3 * 1024);
  k_finalize<256><<<256, 256, 0, stream>>>(zmax, zmin, sums + 3 * 1024, sqs + 3 * 1024,
                                           g4, be4, X4);

  // ---- mlp5 (concat 512 -> 1024) fused with global max pool
  dim3 g5grid(BNW / 64, 4);
  k_mlp5<<<g5grid, 256, 0, stream>>>(X1, X2, X3, X4, Wt5, b5,
                                     sums + 4 * 1024, sqs + 4 * 1024,
                                     (int*)gmax, (int*)gmin);

  // ---- head
  k_head<<<1, 256, 0, stream>>>(sums + 4 * 1024, sqs + 4 * 1024, g5, be5,
                                gmax, gmin,
                                F1W, F1b, F1g, F1be,
                                F2W, F2b, F2g, F2be,
                                OW, Ob, out);
}
